// RecurrentConvBlock_51419348467718
// MI455X (gfx1250) — compile-verified, matrix operand order checked by reading, not run on hardware
//
#include <hip/hip_runtime.h>
#include <hip/hip_bf16.h>

typedef __attribute__((ext_vector_type(16))) __bf16 v16bf;
typedef __attribute__((ext_vector_type(8)))  float  v8f;

#define SA 40    // A-tile LDS row stride (elements), 80B, 16B aligned, bank-spread
#define SB 40    // B-tile LDS row stride
#define SZ 260   // Z-tile LDS row stride (floats)

__device__ __forceinline__ unsigned short f2bf(float f) {
  unsigned u = __float_as_uint(f);
  unsigned r = (u + 0x7FFFu + ((u >> 16) & 1u)) >> 16;   // round-to-nearest-even
  return (unsigned short)r;
}

__device__ __forceinline__ float hsig(float x) {
  return fminf(fmaxf(0.2f * x + 0.5f, 0.0f), 1.0f);
}

// Async global->LDS copy (gfx1250, tracked by ASYNCcnt)
__device__ __forceinline__ void async_ld_b128(unsigned lds_byte_off, const void* gptr) {
  asm volatile("global_load_async_to_lds_b128 %0, %1, off"
               :: "v"(lds_byte_off), "v"(gptr)
               : "memory");
}
__device__ __forceinline__ void wait_async0() {
  asm volatile("s_wait_asynccnt 0" ::: "memory");
}

// ---------------------------------------------------------------------------
// Weight packing: W[k][n] fp32 (k = (dy*kw+dx)*C + c, n fastest in memory)
//   -> bf16 packed as [chunk][n][kk], chunk = k/32, kk = k%32
// so a 32K x 256N B-tile is one contiguous 16KB block.
// ---------------------------------------------------------------------------
__global__ void __launch_bounds__(256) pack_weights(const float* __restrict__ W,
                                                    unsigned short* __restrict__ out,
                                                    int K) {
  int i = blockIdx.x * 256 + threadIdx.x;      // i = k*256 + n
  if (i >= K * 256) return;
  int k = i >> 8;
  int n = i & 255;
  out[(((size_t)(k >> 5) * 256 + n) * 32) + (k & 31)] = f2bf(W[i]);
}

__global__ void __launch_bounds__(256) zero_state(float* __restrict__ c,
                                                  unsigned short* __restrict__ h,
                                                  int n) {
  int i = blockIdx.x * 256 + threadIdx.x;
  if (i < n) { c[i] = 0.0f; h[i] = 0; }
}

// ---------------------------------------------------------------------------
// One ConvLSTM cell step (one layer, one timestep), fully specialized per
// layer via template parameters so all tap decode is constant-folded.
// Grid: 512 blocks x 256 threads (8 waves). Block computes 64 pixels (one
// image row) x 256 gate channels via implicit GEMM over
// K = KH*KH*CIN (input conv) + KH*KH*64 (recurrent conv), chunked by 32.
// A 32-K chunk never straddles a conv tap, so the tap decode is chunk-uniform
// (pure SALU) and each producer thread's 8 K-values are 8 consecutive
// channels of one pixel -> one predicated b128 load.
// Double-buffered LDS: async B-copy / A-gather of chunk n+1 overlaps the 8
// v_wmma_f32_16x16x32_bf16 of chunk n. One barrier per chunk.
// ---------------------------------------------------------------------------
template <int KH, int PAD, int CIN, int ISF, int YF32, int NCHW, int NCHU>
__global__ void __launch_bounds__(256)
convlstm_step(const void* __restrict__ in_act,
              const unsigned short* __restrict__ h_prev,
              const unsigned short* __restrict__ Wp,
              const unsigned short* __restrict__ Up,
              const float* __restrict__ bias,
              float* __restrict__ cbuf,
              unsigned short* __restrict__ h_out,
              void* __restrict__ ybuf,
              int in_t, int in_T, int yt, int yT) {
  __shared__ __align__(16) unsigned short Asm[2][64 * SA];
  __shared__ __align__(16) unsigned short Bsm[2][256 * SB];
  __shared__ __align__(16) float          Zsm[64 * SZ];

  const int tid   = threadIdx.x;
  const int lane  = tid & 31;
  const int wv    = tid >> 5;
  const int mSub  = (wv & 3) * 16;
  const int nBase = (wv >> 2) * 128;
  const int P0    = blockIdx.x * 64;     // first pixel of tile
  const int bImg  = P0 >> 12;            // image index (H*W = 4096)
  const int y0    = (P0 >> 6) & 63;      // row within image

  const int laneN = lane & 15;
  const int koff  = (lane >> 4) * 8;     // ISA 16-bit A/B lane K-offset

  // im2col producer role: 4 threads per pixel row, 8 consecutive channels each
  const int mRow = tid >> 2;             // = x coordinate of pixel
  const int ksA  = (tid & 3) * 8;

  constexpr int TOT = NCHW + NCHU;

  v8f acc[8];
  v8f vzero = {0.f, 0.f, 0.f, 0.f, 0.f, 0.f, 0.f, 0.f};
#pragma unroll
  for (int i = 0; i < 8; ++i) acc[i] = vzero;

  // ---- producer: stage chunk chAll into LDS buffer `buf`
  auto issue_loads = [&](int chAll, int buf) {
    const bool pass = (chAll >= NCHW);
    const int  ch   = pass ? (chAll - NCHW) : chAll;
    // B tile: async-copy 256 rows x 32 bf16 (thread t -> row t)
    {
      const unsigned short* Wpk = pass ? Up : Wp;
      const unsigned short* g = Wpk + ((size_t)ch * 256 + tid) * 32;
      unsigned lb = (unsigned)(unsigned long long)(const void*)&Bsm[buf][tid * SB];
      async_ld_b128(lb + 0,  (const char*)g + 0);
      async_ld_b128(lb + 16, (const char*)g + 16);
      async_ld_b128(lb + 32, (const char*)g + 32);
      async_ld_b128(lb + 48, (const char*)g + 48);
    }
    // A tile: tap is chunk-uniform; 8 consecutive channels per thread
    {
      const int  C    = pass ? 64 : CIN;
      const int  shft = pass ? 1  : (CIN == 64 ? 1 : 0);   // chunks per tap
      const bool isf  = pass ? false : (ISF != 0);
      const int  sT   = pass ? 1 : in_T;
      const int  st   = pass ? 0 : in_t;
      const void* src = pass ? (const void*)h_prev : in_act;

      const int tap   = ch >> shft;                        // uniform
      const int cBase = ((ch & ((1 << shft) - 1)) << 5) + ksA;
      const int dy    = tap / KH;                          // const-divisor SALU
      const int dx    = tap - dy * KH;
      const int yy    = y0 + dy - PAD;                     // uniform
      const int xx    = mRow + dx - PAD;                   // per-thread

      uint4 pk = {0u, 0u, 0u, 0u};
      if ((unsigned)yy < 64u && (unsigned)xx < 64u) {
        size_t base = (((size_t)(bImg * sT + st) * 4096u) +
                       (size_t)(yy * 64 + xx)) * C + cBase;
        if (isf) {
          const float4* f4 = (const float4*)((const float*)src + base);
          float4 a = f4[0], b = f4[1];
          unsigned* pu = (unsigned*)&pk;
          pu[0] = (unsigned)f2bf(a.x) | ((unsigned)f2bf(a.y) << 16);
          pu[1] = (unsigned)f2bf(a.z) | ((unsigned)f2bf(a.w) << 16);
          pu[2] = (unsigned)f2bf(b.x) | ((unsigned)f2bf(b.y) << 16);
          pu[3] = (unsigned)f2bf(b.z) | ((unsigned)f2bf(b.w) << 16);
        } else {
          pk = *(const uint4*)((const unsigned short*)src + base);
        }
      }
      *(uint4*)&Asm[buf][mRow * SA + ksA] = pk;
    }
  };

  // ---- software pipeline: prologue stages chunk 0
  issue_loads(0, 0);

  for (int chAll = 0; chAll < TOT; ++chAll) {
    const int buf = chAll & 1;
    wait_async0();        // my async B-copies for chunk chAll complete
    __syncthreads();      // whole tile (A ds_store + B async) published
    if (chAll + 1 < TOT) issue_loads(chAll + 1, buf ^ 1);

    // fragment gathers + 8x v_wmma_f32_16x16x32_bf16 per wave
    union { uint4 u[2]; v16bf v; } af, bfr;
    af.u[0] = *(const uint4*)&Asm[buf][(mSub + laneN) * SA + koff];
    af.u[1] = *(const uint4*)&Asm[buf][(mSub + laneN) * SA + 16 + koff];
#pragma unroll
    for (int nt = 0; nt < 8; ++nt) {
      int nr = nBase + nt * 16 + laneN;
      bfr.u[0] = *(const uint4*)&Bsm[buf][nr * SB + koff];
      bfr.u[1] = *(const uint4*)&Bsm[buf][nr * SB + 16 + koff];
      acc[nt] = __builtin_amdgcn_wmma_f32_16x16x32_bf16(
          false, af.v, false, bfr.v, (short)0, acc[nt], false, false);
    }
  }

  // ---- spill accumulators to LDS z-tile (C/D layout: lane>=16 -> rows +8)
  __syncthreads();
  {
    int mOff = mSub + koff;              // koff == (lane>>4)*8
#pragma unroll
    for (int nt = 0; nt < 8; ++nt) {
      int n = nBase + nt * 16 + laneN;
#pragma unroll
      for (int r = 0; r < 8; ++r)
        Zsm[(mOff + r) * SZ + n] = acc[nt][r];
    }
  }
  __syncthreads();

  // ---- fused LSTM elementwise update (gate order i,f,c,o)
  for (int i = 0; i < 16; ++i) {
    int idx = tid + i * 256;             // over 64 pixels x 64 channels
    int pl  = idx >> 6;
    int f0  = idx & 63;
    float zi = Zsm[pl * SZ + f0]        + bias[f0];
    float zf = Zsm[pl * SZ + 64 + f0]   + bias[64 + f0];
    float zc = Zsm[pl * SZ + 128 + f0]  + bias[128 + f0];
    float zo = Zsm[pl * SZ + 192 + f0]  + bias[192 + f0];
    float ig = hsig(zi), fg = hsig(zf), og = hsig(zo);
    size_t pc = (size_t)(P0 + pl) * 64 + f0;
    float cn = fg * cbuf[pc] + ig * tanhf(zc);
    cbuf[pc] = cn;
    float h = og * tanhf(cn);
    h_out[pc] = f2bf(h);
    float yv = fmaxf(h, 0.0f);
    int s = (P0 + pl) & 4095;
    size_t yi = ((size_t)(bImg * yT + yt) * 4096 + s) * 64 + f0;
    if (YF32) ((float*)ybuf)[yi] = yv;
    else      ((unsigned short*)ybuf)[yi] = f2bf(yv);
  }
}

// ---------------------------------------------------------------------------
extern "C" void kernel_launch(void* const* d_in, const int* in_sizes, int n_in,
                              void* d_out, int out_size, void* d_ws, size_t ws_size,
                              hipStream_t stream) {
  const float* x   = (const float*)d_in[0];
  const float* Wk1 = (const float*)d_in[1];
  const float* Uk1 = (const float*)d_in[2];
  const float* b1  = (const float*)d_in[3];
  const float* Wk2 = (const float*)d_in[4];
  const float* Uk2 = (const float*)d_in[5];
  const float* b2  = (const float*)d_in[6];
  float* out = (float*)d_out;

  const size_t P  = (size_t)8 * 4096;    // 32768 pixels
  const size_t SN = P * 64;              // state elements

  char* ws = (char*)d_ws;
  auto take = [&](size_t bytes) {
    char* p = ws;
    ws += (bytes + 255) & ~(size_t)255;
    return p;
  };
  unsigned short* W1p = (unsigned short*)take((size_t)800  * 256 * 2);
  unsigned short* U1p = (unsigned short*)take((size_t)1600 * 256 * 2);
  unsigned short* W2p = (unsigned short*)take((size_t)576  * 256 * 2);
  unsigned short* U2p = (unsigned short*)take((size_t)576  * 256 * 2);
  unsigned short* h1a = (unsigned short*)take(SN * 2);
  unsigned short* h1b = (unsigned short*)take(SN * 2);
  unsigned short* h2a = (unsigned short*)take(SN * 2);
  unsigned short* h2b = (unsigned short*)take(SN * 2);
  float*          c1  = (float*)take(SN * 4);
  float*          c2  = (float*)take(SN * 4);
  unsigned short* y1  = (unsigned short*)take(SN * 2);

  pack_weights<<<(800  * 256) / 256, 256, 0, stream>>>(Wk1, W1p, 800);
  pack_weights<<<(1600 * 256) / 256, 256, 0, stream>>>(Uk1, U1p, 1600);
  pack_weights<<<(576  * 256) / 256, 256, 0, stream>>>(Wk2, W2p, 576);
  pack_weights<<<(576  * 256) / 256, 256, 0, stream>>>(Uk2, U2p, 576);
  zero_state<<<(int)((SN + 255) / 256), 256, 0, stream>>>(c1, h1a, (int)SN);
  zero_state<<<(int)((SN + 255) / 256), 256, 0, stream>>>(c2, h2a, (int)SN);

  for (int t = 0; t < 10; ++t) {
    unsigned short* h1p = (t & 1) ? h1b : h1a;
    unsigned short* h1n = (t & 1) ? h1a : h1b;
    unsigned short* h2p = (t & 1) ? h2b : h2a;
    unsigned short* h2n = (t & 1) ? h2a : h2b;
    // layer 1: x_t (fp32, strided by t) -> y1 (bf16)
    convlstm_step<5, 2, 32, 1, 0, 25, 50><<<512, 256, 0, stream>>>(
        x, h1p, W1p, U1p, b1, c1, h1n,
        y1, /*in_t*/t, /*in_T*/10, /*yt*/0, /*yT*/1);
    // layer 2: y1 (bf16) -> out slice t (fp32)
    convlstm_step<3, 1, 64, 0, 1, 18, 18><<<512, 256, 0, stream>>>(
        y1, h2p, W2p, U2p, b2, c2, h2n,
        out, /*in_t*/0, /*in_T*/1, /*yt*/t, /*yT*/10);
  }
  (void)in_sizes; (void)n_in; (void)out_size; (void)ws_size;
}
